// DS4_16801912062573
// MI455X (gfx1250) — matrix operand order repaired and checked by reading, not compiled, not run on hardware
//
#include <hip/hip_runtime.h>
#include <math.h>

typedef __attribute__((ext_vector_type(16))) __bf16 bf16x16;
typedef __attribute__((ext_vector_type(8)))  __bf16 bf16x8;
typedef __attribute__((ext_vector_type(8)))  float  f32x8;

__device__ __forceinline__ float gelu_exact(float x) {
  return 0.5f * x * (1.0f + erff(x * 0.7071067811865476f));
}

__device__ __forceinline__ bf16x16 concat8(bf16x8 lo, bf16x8 hi) {
  return __builtin_shufflevector(lo, hi, 0, 1, 2, 3, 4, 5, 6, 7,
                                 8, 9, 10, 11, 12, 13, 14, 15);
}

// ---------------- elementwise f32 -> bf16 (optional gelu) ----------------
__global__ void k_f32_to_bf16(const float* __restrict__ src, __bf16* __restrict__ dst,
                              int n, int act) {
  int i = blockIdx.x * blockDim.x + threadIdx.x;
  if (i >= n) return;
  float v = src[i];
  if (act == 2) v = gelu_exact(v);
  dst[i] = (__bf16)v;
}

// ---------------- pack weights into WMMA B-fragment order ----------------
// P layout: [nt][kp][lane][e] contiguous; element (lane,e) -> k = kp*32 + 8*(lane>>4) + (e<8 ? e : e+8),
// n = nt*16 + (lane&15). Zero-pads n >= N (Npad = padded N, multiple of 16).
// transposed==0: W is [K,N] row-major; transposed==1: W is [N,K] row-major.
__global__ void k_packB(const float* __restrict__ W, __bf16* __restrict__ P,
                        int K, int N, int Npad, int transposed) {
  int tid = blockIdx.x * blockDim.x + threadIdx.x;
  int total = Npad * K;
  if (tid >= total) return;
  int e = tid & 15;
  int lane = (tid >> 4) & 31;
  int rest = tid >> 9;
  int kpc = K >> 5;
  int kp = rest % kpc;
  int nt = rest / kpc;
  int k = (kp << 5) + ((lane >> 4) << 3) + (e < 8 ? e : e + 8);
  int n = (nt << 4) + (lane & 15);
  float v = 0.f;
  if (n < N) v = transposed ? W[(size_t)n * K + k] : W[(size_t)k * N + n];
  P[tid] = (__bf16)v;
}

// ---------------- WMMA GEMM: C = act(A@Bp + bias) (+ residual) ----------------
// Requires M % (MT*16) == 0, K % 32 == 0, Bp packed with Npad = gridDim.x*NT*16.
// One wave computes (MT*16) x (NT*16). act: 0=none,1=tanh,2=gelu.
template <int MT, int NT>
__global__ __launch_bounds__(32)
void k_gemm_wmma(const __bf16* __restrict__ A, int lda,
                 const __bf16* __restrict__ Bp, int K,
                 const float* __restrict__ bias,
                 const float* res, int ldres,
                 float* C, int ldc,
                 int M, int N, int act) {
  const int lane = threadIdx.x;
  const int half = lane >> 4, l16 = lane & 15;
  const int m0 = blockIdx.y * (MT * 16);
  const int nt0 = blockIdx.x * NT;
  const int kpc = K >> 5;

  const f32x8 zero = {0.f, 0.f, 0.f, 0.f, 0.f, 0.f, 0.f, 0.f};
  f32x8 acc[MT][NT];
#pragma unroll
  for (int mi = 0; mi < MT; ++mi)
#pragma unroll
    for (int ni = 0; ni < NT; ++ni) acc[mi][ni] = zero;

  const __bf16* ap[MT];
#pragma unroll
  for (int mi = 0; mi < MT; ++mi)
    ap[mi] = A + (size_t)(m0 + mi * 16 + l16) * lda + (half << 3);
  const __bf16* bp0 = Bp + ((size_t)nt0 * kpc * 32 + lane) * 16;

  for (int kp = 0; kp < kpc; ++kp) {
    if (kp + 2 < kpc) __builtin_prefetch(ap[0] + (kp + 2) * 32, 0, 3);
    bf16x16 afr[MT], bfr[NT];
#pragma unroll
    for (int mi = 0; mi < MT; ++mi) {
      bf16x8 lo = *(const bf16x8*)(ap[mi] + kp * 32);
      bf16x8 hi = *(const bf16x8*)(ap[mi] + kp * 32 + 16);
      afr[mi] = concat8(lo, hi);
    }
#pragma unroll
    for (int ni = 0; ni < NT; ++ni) {
      const bf16x8* bp = (const bf16x8*)(bp0 + ((size_t)ni * kpc + kp) * 32 * 16);
      bfr[ni] = concat8(bp[0], bp[1]);
    }
#pragma unroll
    for (int mi = 0; mi < MT; ++mi)
#pragma unroll
      for (int ni = 0; ni < NT; ++ni)
        acc[mi][ni] = __builtin_amdgcn_wmma_f32_16x16x32_bf16(
            false, afr[mi], false, bfr[ni], (short)0, acc[mi][ni], false, false);
  }

#pragma unroll
  for (int mi = 0; mi < MT; ++mi) {
#pragma unroll
    for (int ni = 0; ni < NT; ++ni) {
      int n = (nt0 + ni) * 16 + l16;
      if (n < N) {
#pragma unroll
        for (int r = 0; r < 8; ++r) {
          int m = m0 + mi * 16 + (half << 3) + r;
          float v = acc[mi][ni][r];
          if (bias) v += bias[n];
          if (act == 1) v = tanhf(v);
          else if (act == 2) v = gelu_exact(v);
          if (res) v += res[(size_t)m * ldres + n];
          C[(size_t)m * ldc + n] = v;
        }
      }
    }
  }
}

// ---------------- implicit-GEMM conv (VALID, square kernel) + ReLU ----------------
// M = nframes*OH*OW (multiple of 16), Ktot = Cin*KS*KS (multiple of 32), OC = NT*16.
// A gathered from f32 input on the fly (branch-free); weights pre-packed bf16.
template <int NT>
__global__ __launch_bounds__(32)
void k_conv_wmma(const float* __restrict__ in,
                 long base_off, long batch_stride, long frame_stride, int fpb,
                 const __bf16* __restrict__ Wp, const float* __restrict__ bias,
                 float* __restrict__ out,
                 int Cin, int IH, int IW, int KS, int stride,
                 int OC, int OH, int OW) {
  const int KHW = KS * KS, Ktot = Cin * KHW, OHW = OH * OW;
  const int lane = threadIdx.x;
  const int half = lane >> 4, l16 = lane & 15;
  const int m0 = blockIdx.y << 4;
  const int nt0 = blockIdx.x * NT;
  const int kpc = Ktot >> 5;

  const int am = m0 + l16;
  const int frame = am / OHW;
  const int p = am - frame * OHW;
  const int oy = p / OW, ox = p - oy * OW;
  const int iy0 = oy * stride, ix0 = ox * stride;
  const long inbase = base_off + (long)(frame / fpb) * batch_stride +
                      (long)(frame % fpb) * frame_stride;

  const f32x8 zero = {0.f, 0.f, 0.f, 0.f, 0.f, 0.f, 0.f, 0.f};
  f32x8 acc[NT];
#pragma unroll
  for (int ni = 0; ni < NT; ++ni) acc[ni] = zero;
  const __bf16* bp0 = Wp + ((size_t)nt0 * kpc * 32 + lane) * 16;

  for (int kp = 0; kp < kpc; ++kp) {
    bf16x16 a, bfr[NT];
#pragma unroll
    for (int e = 0; e < 16; ++e) {
      int ka = (kp << 5) + (half << 3) + (e < 8 ? e : e + 8);
      int ic = ka / KHW;
      int rem = ka - ic * KHW;
      int ky = rem / KS;
      int kx = rem - ky * KS;
      a[e] = (__bf16)in[inbase + ((long)ic * IH + (iy0 + ky)) * IW + (ix0 + kx)];
    }
#pragma unroll
    for (int ni = 0; ni < NT; ++ni) {
      const bf16x8* bp = (const bf16x8*)(bp0 + ((size_t)ni * kpc + kp) * 32 * 16);
      bfr[ni] = concat8(bp[0], bp[1]);
    }
#pragma unroll
    for (int ni = 0; ni < NT; ++ni)
      acc[ni] = __builtin_amdgcn_wmma_f32_16x16x32_bf16(
          false, a, false, bfr[ni], (short)0, acc[ni], false, false);
  }

#pragma unroll
  for (int ni = 0; ni < NT; ++ni) {
    int n = (nt0 + ni) * 16 + l16;
#pragma unroll
    for (int r = 0; r < 8; ++r) {
      int m = m0 + (half << 3) + r;
      float v = fmaxf(acc[ni][r] + bias[n], 0.f);
      int fr = m / OHW, pp = m - fr * OHW;
      out[((long)fr * OC + n) * OHW + pp] = v;  // [frame][oc][oy][ox]
    }
  }
}

// ---------------- action / return embeddings into u384 cols 128..383 ----------------
__global__ void k_embed(const int* __restrict__ actions, const float* __restrict__ rtgs,
                        const float* __restrict__ act_emb, const float* __restrict__ ret_w,
                        const float* __restrict__ ret_b, float* __restrict__ u384) {
  int tid = blockIdx.x * blockDim.x + threadIdx.x;  // 1024*128
  if (tid >= 1024 * 128) return;
  int i = tid >> 7, j = tid & 127;
  int b = i >> 7, s = i & 127;
  int a = actions[b * 129 + s];
  u384[(size_t)i * 384 + 128 + j] = tanhf(act_emb[a * 128 + j]);
  float rv = rtgs[b * 129 + s];
  u384[(size_t)i * 384 + 256 + j] = tanhf(rv * ret_w[j] + ret_b[j]);
}

// ---------------- layernorm over H=512, one wave per row ----------------
__global__ __launch_bounds__(32)
void k_layernorm(const float* __restrict__ y, const float* __restrict__ g,
                 const float* __restrict__ b, float* __restrict__ r) {
  int row = blockIdx.x;
  int lane = threadIdx.x;
  const float* yr = y + (size_t)row * 512;
  float v[16];
  float s1 = 0.f, s2 = 0.f;
#pragma unroll
  for (int j = 0; j < 16; ++j) {
    v[j] = yr[lane + 32 * j];
    s1 += v[j];
    s2 += v[j] * v[j];
  }
#pragma unroll
  for (int off = 16; off > 0; off >>= 1) {
    s1 += __shfl_xor(s1, off, 32);
    s2 += __shfl_xor(s2, off, 32);
  }
  float mu = s1 * (1.f / 512.f);
  float var = s2 * (1.f / 512.f) - mu * mu;
  float rs = rsqrtf(var + 1e-5f);
  float* rr = r + (size_t)row * 512;
#pragma unroll
  for (int j = 0; j < 16; ++j) {
    int hh = lane + 32 * j;
    rr[hh] = (v[j] - mu) * rs * g[hh] + b[hh];
  }
}

// ---------------- S4 spectral kernel: K[h,l] = 2*Re(sum_n Ceff * exp(dtA)^l) ----------------
__global__ void k_s4kernel(const float* __restrict__ ldt, const float* __restrict__ Ar,
                           const float* __restrict__ Ai, const float* __restrict__ Cr,
                           const float* __restrict__ Ci, float* __restrict__ Kout) {
  int tid = blockIdx.x * blockDim.x + threadIdx.x;  // 512*128
  if (tid >= 512 * 128) return;
  int h = tid >> 7, l = tid & 127;
  float dt = expf(ldt[h]);
  float fl = (float)l;
  float acc = 0.f;
  for (int n = 0; n < 64; ++n) {
    float ar = Ar[h * 64 + n], ai = Ai[h * 64 + n];
    float dr = dt * ar, di = dt * ai;
    float m0 = expf(dr);
    float nr = m0 * cosf(di) - 1.0f;  // exp(dtA) - 1
    float ni = m0 * sinf(di);
    float den = ar * ar + ai * ai;
    float tr = (nr * ar + ni * ai) / den;  // (exp(dtA)-1)/A
    float ti = (ni * ar - nr * ai) / den;
    float cr = Cr[h * 64 + n], ci = Ci[h * 64 + n];
    float gr = cr * tr - ci * ti;  // Ceff
    float gi = cr * ti + ci * tr;
    float mg = expf(dr * fl);
    float ph = di * fl;
    float er = mg * cosf(ph), ei = mg * sinf(ph);  // exp(dtA)^l
    acc += gr * er - gi * ei;
  }
  Kout[h * 128 + l] = 2.0f * acc;
}

// ---------------- causal Toeplitz conv along time + skip D*u ----------------
__global__ void k_causal(const float* __restrict__ r, const float* __restrict__ Kf,
                         const float* __restrict__ Dp, float* __restrict__ c) {
  int tid = blockIdx.x * blockDim.x + threadIdx.x;  // 1024*512
  if (tid >= 1024 * 512) return;
  int h = tid & 511, row = tid >> 9;
  int s = row & 127;
  const float* Kh = Kf + h * 128;
  long base = (long)(row - s) * 512 + h;
  float acc = Dp[h] * r[(size_t)row * 512 + h];
  for (int j = 0; j <= s; ++j)
    acc = fmaf(Kh[j], r[base + (long)(s - j) * 512], acc);
  c[(size_t)row * 512 + h] = acc;
}

extern "C" void kernel_launch(void* const* d_in, const int* in_sizes, int n_in,
                              void* d_out, int out_size, void* d_ws, size_t ws_size,
                              hipStream_t stream) {
  const float* states  = (const float*)d_in[0];
  const int*   actions = (const int*)d_in[1];
  const float* rtgs    = (const float*)d_in[2];
  const float* w1      = (const float*)d_in[3];
  const float* b1      = (const float*)d_in[4];
  const float* w2      = (const float*)d_in[5];
  const float* b2      = (const float*)d_in[6];
  const float* w3      = (const float*)d_in[7];
  const float* b3      = (const float*)d_in[8];
  const float* enc_lw  = (const float*)d_in[9];
  const float* enc_lb  = (const float*)d_in[10];
  const float* ret_w   = (const float*)d_in[11];
  const float* ret_b   = (const float*)d_in[12];
  const float* act_emb = (const float*)d_in[13];
  const float* proj_w  = (const float*)d_in[14];
  const float* proj_b  = (const float*)d_in[15];
  const float* pre_w   = (const float*)d_in[16];
  const float* pre_b   = (const float*)d_in[17];
  const float* ln_g    = (const float*)d_in[18];
  const float* ln_b    = (const float*)d_in[19];
  const float* log_dt  = (const float*)d_in[20];
  const float* A_re    = (const float*)d_in[21];
  const float* A_im    = (const float*)d_in[22];
  const float* C_re    = (const float*)d_in[23];
  const float* C_im    = (const float*)d_in[24];
  const float* Dp      = (const float*)d_in[25];
  const float* aft_w   = (const float*)d_in[26];
  const float* aft_b   = (const float*)d_in[27];
  const float* out_w   = (const float*)d_in[28];
  float* logits = (float*)d_out;

  char* wp = (char*)d_ws;
  size_t off = 0;
  auto alloc = [&](size_t bytes) -> char* {
    char* p = wp + off;
    off += (bytes + 255) & ~(size_t)255;
    return p;
  };
  __bf16* wc1p  = (__bf16*)alloc((size_t)256 * 32 * 2);
  __bf16* wc2p  = (__bf16*)alloc((size_t)512 * 64 * 2);
  __bf16* wc3p  = (__bf16*)alloc((size_t)576 * 64 * 2);
  __bf16* lwp   = (__bf16*)alloc((size_t)3136 * 128 * 2);
  __bf16* pjp   = (__bf16*)alloc((size_t)384 * 512 * 2);
  __bf16* prp   = (__bf16*)alloc((size_t)512 * 512 * 2);
  __bf16* afp   = (__bf16*)alloc((size_t)4 * 512 * 512 * 2);
  __bf16* owp   = (__bf16*)alloc((size_t)512 * 32 * 2);   // N=18 padded to 32
  float*  u384f = (float*)alloc((size_t)1024 * 384 * 4);
  __bf16* u384b = (__bf16*)alloc((size_t)1024 * 384 * 2);
  float*  ubuf  = (float*)alloc((size_t)1024 * 512 * 4);
  __bf16* actb  = (__bf16*)alloc((size_t)1024 * 512 * 2);
  float*  ybuf  = (float*)alloc((size_t)1024 * 512 * 4);
  float*  rbuf  = (float*)alloc((size_t)1024 * 512 * 4);
  float*  cbuf  = (float*)alloc((size_t)1024 * 512 * 4);
  float*  Kbuf  = (float*)alloc((size_t)512 * 128 * 4);
  float*  c2out = (float*)alloc((size_t)1024 * 64 * 81 * 4);
  float*  c1out = (float*)alloc((size_t)1024 * 32 * 400 * 4);  // big region
  float*  c3out = c1out;                                       // reuse (conv1 out dead)
  __bf16* c3bf  = (__bf16*)((char*)c1out + (size_t)16 * 1024 * 1024);

  dim3 b256(256), wave(32);

  // pack weights into WMMA B-fragment order (bf16)
  k_packB<<<(32 * 256 + 255) / 256, b256, 0, stream>>>(w1, wc1p, 256, 32, 32, 1);
  k_packB<<<(64 * 512 + 255) / 256, b256, 0, stream>>>(w2, wc2p, 512, 64, 64, 1);
  k_packB<<<(64 * 576 + 255) / 256, b256, 0, stream>>>(w3, wc3p, 576, 64, 64, 1);
  k_packB<<<(128 * 3136 + 255) / 256, b256, 0, stream>>>(enc_lw, lwp, 3136, 128, 128, 0);
  k_packB<<<(512 * 384 + 255) / 256, b256, 0, stream>>>(proj_w, pjp, 384, 512, 512, 0);
  k_packB<<<(512 * 512 + 255) / 256, b256, 0, stream>>>(pre_w, prp, 512, 512, 512, 0);
  for (int i = 0; i < 4; ++i)
    k_packB<<<(512 * 512 + 255) / 256, b256, 0, stream>>>(
        aft_w + (size_t)i * 512 * 512, afp + (size_t)i * 512 * 512, 512, 512, 512, 0);
  k_packB<<<(32 * 512 + 255) / 256, b256, 0, stream>>>(out_w, owp, 512, 18, 32, 0);

  // CNN encoder over 1024 frames: states[:,1:] -> 3x conv+relu (implicit-GEMM WMMA)
  k_conv_wmma<2><<<dim3(1, 409600 / 16), wave, 0, stream>>>(
      states, 28224L, 3640896L, 28224L, 128, wc1p, b1, c1out,
      4, 84, 84, 8, 4, 32, 20, 20);
  k_conv_wmma<4><<<dim3(1, 82944 / 16), wave, 0, stream>>>(
      c1out, 0L, 0L, 12800L, 1 << 30, wc2p, b2, c2out,
      32, 20, 20, 4, 2, 64, 9, 9);
  k_conv_wmma<4><<<dim3(1, 50176 / 16), wave, 0, stream>>>(
      c2out, 0L, 0L, 5184L, 1 << 30, wc3p, b3, c3out,
      64, 9, 9, 3, 1, 64, 7, 7);
  k_f32_to_bf16<<<(3211264 + 255) / 256, b256, 0, stream>>>(c3out, c3bf, 3211264, 0);

  // se = tanh(flat @ enc_lw + enc_lb) -> u384 cols [0,128)
  k_gemm_wmma<2, 4><<<dim3(2, 32), wave, 0, stream>>>(
      c3bf, 3136, lwp, 3136, enc_lb, nullptr, 0, u384f, 384, 1024, 128, 1);
  // ae / re -> u384 cols [128,384)
  k_embed<<<(1024 * 128) / 256, b256, 0, stream>>>(actions, rtgs, act_emb, ret_w, ret_b, u384f);
  k_f32_to_bf16<<<(1024 * 384 + 255) / 256, b256, 0, stream>>>(u384f, u384b, 1024 * 384, 0);

  // u = u384 @ proj_w + proj_b
  k_gemm_wmma<2, 4><<<dim3(8, 32), wave, 0, stream>>>(
      u384b, 384, pjp, 384, proj_b, nullptr, 0, ubuf, 512, 1024, 512, 0);
  // y = gelu(u) @ pre_w + pre_b
  k_f32_to_bf16<<<(1024 * 512 + 255) / 256, b256, 0, stream>>>(ubuf, actb, 1024 * 512, 2);
  k_gemm_wmma<2, 4><<<dim3(8, 32), wave, 0, stream>>>(
      actb, 512, prp, 512, pre_b, nullptr, 0, ybuf, 512, 1024, 512, 0);

  // 4 S4 residual blocks
  for (int i = 0; i < 4; ++i) {
    k_layernorm<<<1024, wave, 0, stream>>>(ybuf, ln_g + i * 512, ln_b + i * 512, rbuf);
    k_s4kernel<<<(512 * 128) / 256, b256, 0, stream>>>(
        log_dt + i * 512, A_re + (size_t)i * 512 * 64, A_im + (size_t)i * 512 * 64,
        C_re + (size_t)i * 512 * 64, C_im + (size_t)i * 512 * 64, Kbuf);
    k_causal<<<(1024 * 512) / 256, b256, 0, stream>>>(rbuf, Kbuf, Dp + i * 512, cbuf);
    k_f32_to_bf16<<<(1024 * 512 + 255) / 256, b256, 0, stream>>>(cbuf, actb, 1024 * 512, 2);
    // y = gelu(gelu(c) @ aft_w + aft_b) + y  (in-place residual: read-then-write per elem)
    k_gemm_wmma<2, 4><<<dim3(8, 32), wave, 0, stream>>>(
        actb, 512, afp + (size_t)i * 512 * 512, 512, aft_b + i * 512,
        ybuf, 512, ybuf, 512, 1024, 512, 2);
  }

  // logits = y @ out_w  (B packed/padded to N=32; stores guarded to n<18)
  k_f32_to_bf16<<<(1024 * 512 + 255) / 256, b256, 0, stream>>>(ybuf, actb, 1024 * 512, 0);
  k_gemm_wmma<2, 2><<<dim3(1, 32), wave, 0, stream>>>(
      actb, 512, owp, 512, nullptr, nullptr, 0, logits, 18, 1024, 18, 0);
}
